// SimpleLSTMCell_39848706573452
// MI455X (gfx1250) — matrix operand order repaired and checked by reading, not compiled, not run on hardware
//
#include <hip/hip_runtime.h>
#include <hip/hip_bf16.h>

#define BROWS 131072
#define HD    256
#define IND   3
#define DD    259      // IN + H
#define KP    288      // K padded to multiple of 32

typedef __attribute__((ext_vector_type(16))) __bf16 bfx16;
typedef __attribute__((ext_vector_type(8)))  float  v8f;

union Frag { uint4 q[2]; bfx16 v; };

__device__ __forceinline__ unsigned short f2bf(float f) {
    unsigned u = __float_as_uint(f);
    unsigned r = u + 0x7FFFu + ((u >> 16) & 1u);   // round-to-nearest-even
    return (unsigned short)(r >> 16);
}
__device__ __forceinline__ float bf2f(unsigned short h) {
    return __uint_as_float(((unsigned)h) << 16);
}
__device__ __forceinline__ float sigmoidf(float x) {
    return 1.0f / (1.0f + __expf(-x));
}
__device__ __forceinline__ float tanh_fast(float x) {
    float e = __expf(-2.0f * fabsf(x));
    float r = (1.0f - e) / (1.0f + e);
    return copysignf(r, x);
}

// ---------------- pass 0: zero counters ----------------
__global__ void zero_cnt_k(int* c) {
    if (threadIdx.x < 2) c[threadIdx.x] = 0;
}

// ---------------- pass 1: W_gates f32 (D,1024) -> bf16 [1024][KP] (transposed, padded) ----
__global__ void prep_w_k(const float* __restrict__ Wg, unsigned short* __restrict__ Wbf) {
    int i = blockIdx.x * 256 + threadIdx.x;
    if (i >= 1024 * KP) return;
    int col = i / KP;
    int k   = i % KP;
    float v = (k < DD) ? Wg[(size_t)k * 1024 + col] : 0.0f;
    Wbf[i] = f2bf(v);
}

// ---------------- pass 2: combined = [input | h_cur] -> bf16 [B][KP], zero padded ---------
__global__ void prep_combined_k(const float* __restrict__ in,
                                const float* __restrict__ h,
                                unsigned short* __restrict__ cb) {
    const long total = (long)BROWS * KP;
    for (long i = (long)blockIdx.x * 256 + threadIdx.x; i < total; i += (long)gridDim.x * 256) {
        long row = i / KP;
        int  k   = (int)(i % KP);
        float v = 0.0f;
        if (k < IND)      v = in[row * IND + k];
        else if (k < DD)  v = h[row * HD + (k - IND)];
        cb[i] = f2bf(v);
    }
}

// ---------------- pass 3: switch GEMV + global counts (sigmoid(x)>0.5 <=> x>0) ------------
__global__ void switch_count_k(const unsigned short* __restrict__ cb,
                               const float* __restrict__ Wsw,
                               const float* __restrict__ bsw,
                               int* __restrict__ cnt) {
    __shared__ int sI, sG;
    if (threadIdx.x == 0) { sI = 0; sG = 0; }
    __syncthreads();
    int row = blockIdx.x * 256 + threadIdx.x;
    float s0 = bsw[0], s1 = bsw[1];
    const unsigned short* rp = cb + (size_t)row * KP;
    for (int k = 0; k < DD; ++k) {
        float x = bf2f(rp[k]);
        s0 = fmaf(x, Wsw[k * 2 + 0], s0);
        s1 = fmaf(x, Wsw[k * 2 + 1], s1);
    }
    if (s0 > 0.0f) atomicAdd(&sI, 1);
    if (s1 > 0.0f) atomicAdd(&sG, 1);
    __syncthreads();
    if (threadIdx.x == 0) { atomicAdd(&cnt[0], sI); atomicAdd(&cnt[1], sG); }
}

// ---------------- pass 4: WMMA GEMM + fused LSTM epilogue --------------------------------
// block = 256 threads (8 waves), owns 64 batch rows.
// wave w: m16 = w&3 (16-row subtile), cgi = w>>2 (which of 2 column-groups this iter).
// Each wave keeps 4 f32 accumulators (gates i,f,g,o) for its 16x16 tile.
__global__ void lstm_wmma_k(const unsigned short* __restrict__ Wbf,
                            const unsigned short* __restrict__ Cbf,
                            const float* __restrict__ bg,
                            const float* __restrict__ c_cur,
                            const int*  __restrict__ cnt,
                            float* __restrict__ out) {
    extern __shared__ unsigned short smem[];
    unsigned short* At = smem;              // [64][KP]  bf16
    unsigned short* Wt = smem + 64 * KP;    // [128][KP] bf16 (cgi*4+g)*16+n major

    const int tid  = threadIdx.x;
    const int lane = tid & 31;
    const int wv   = tid >> 5;
    const int m16  = wv & 3;
    const int cgi  = wv >> 2;                  // 0..1
    const int rowBase = blockIdx.x * 64;

    // ---- load A tile once (uint4 = 8 bf16 per chunk; KP/8 = 36 chunks/row) ----
    for (int li = tid; li < 64 * 36; li += 256) {
        int row = li / 36, ch = li % 36;
        ((uint4*)(At + row * KP))[ch] =
            ((const uint4*)(Cbf + (size_t)(rowBase + row) * KP))[ch];
    }

    const int cI = cnt[0], cG = cnt[1];        // thr = B/2 = 65536 exactly
    const bool condI = (cG < 65536) && (cI > 65536);
    const bool condG = (cG > 65536) && (cI < 65536);

    const int n    = lane & 15;
    const int mrow = m16 * 16 + (lane & 15);   // A-frag source row (ISA 16-bit A layout)
    const int aoff = (lane < 16) ? 0 : 8;      // lanes 16-31 hold K {8-15, 24-31}
    const int boff = (lane < 16) ? 0 : 8;      // B: lanes 16-31 hold K {8-15, 24-31}
    const int rsub = (lane >> 4) * 8;          // C layout: VGPR r -> M = r + 8*(lane>=16)

    for (int it = 0; it < 8; ++it) {
        __syncthreads();
        // ---- load W tile: 128 columns = [cgi(2)][gate(4)][n(16)], 36 uint4 each ----
        for (int li = tid; li < 128 * 36; li += 256) {
            int colIdx = li / 36, ch = li % 36;
            int lcgi = colIdx >> 6;
            int g    = (colIdx >> 4) & 3;
            int ln   = colIdx & 15;
            int col  = g * 256 + (it * 2 + lcgi) * 16 + ln;
            ((uint4*)(Wt + colIdx * KP))[ch] =
                ((const uint4*)(Wbf + (size_t)col * KP))[ch];
        }
        __syncthreads();

        const int c0 = (it * 2 + cgi) * 16;    // column group within H=256

        v8f acc[4];
        #pragma unroll
        for (int g = 0; g < 4; ++g) {
            float bias = bg[g * 256 + c0 + n]; // fold bias into accumulator init
            #pragma unroll
            for (int r = 0; r < 8; ++r) acc[g][r] = bias;
        }

        #pragma unroll
        for (int ks = 0; ks < 9; ++ks) {
            const int k0 = ks * 32;
            Frag a;
            const unsigned short* ap = At + mrow * KP + k0 + aoff;
            a.q[0] = *(const uint4*)ap;            // K {0-7}   (or {8-15})
            a.q[1] = *(const uint4*)(ap + 16);     // K {16-23} (or {24-31})
            #pragma unroll
            for (int g = 0; g < 4; ++g) {
                Frag b;
                const unsigned short* bp = Wt + ((cgi * 4 + g) * 16 + n) * KP + k0 + boff;
                b.q[0] = *(const uint4*)bp;        // K {0-7}   (or {8-15})
                b.q[1] = *(const uint4*)(bp + 16); // K {16-23} (or {24-31})
                acc[g] = __builtin_amdgcn_wmma_f32_16x16x32_bf16(
                    false, a.v, false, b.v, (short)0, acc[g], false, false);
            }
        }

        // ---- fused LSTM epilogue (wave-local: all 4 gates in registers) ----
        const int col = c0 + n;
        #pragma unroll
        for (int r = 0; r < 8; ++r) {
            int grow = rowBase + m16 * 16 + rsub + r;
            float iv = sigmoidf(acc[0][r]);
            float fv = sigmoidf(acc[1][r]);
            float gv = tanh_fast(acc[2][r]);
            float ov = sigmoidf(acc[3][r]);
            float cc = c_cur[(size_t)grow * HD + col];
            float fc = fv * cc;
            float cn = condI ? (fc + iv) : (condG ? (fc + gv) : (fc + iv * gv));
            float hn = ov * tanh_fast(cn);
            out[(size_t)grow * HD + col] = hn;                         // h_next
            out[(size_t)BROWS * HD + (size_t)grow * HD + col] = cn;    // c_next
        }
    }
}

extern "C" void kernel_launch(void* const* d_in, const int* in_sizes, int n_in,
                              void* d_out, int out_size, void* d_ws, size_t ws_size,
                              hipStream_t stream) {
    const float* input    = (const float*)d_in[0];
    const float* h_cur    = (const float*)d_in[1];
    const float* c_cur    = (const float*)d_in[2];
    const float* W_gates  = (const float*)d_in[3];
    const float* b_gates  = (const float*)d_in[4];
    const float* W_switch = (const float*)d_in[5];
    const float* b_switch = (const float*)d_in[6];
    float* out = (float*)d_out;

    char* ws = (char*)d_ws;
    int* cnt = (int*)ws;                                     // 8 B counters
    unsigned short* Wbf = (unsigned short*)(ws + 256);       // 1024*288*2 = 576 KB
    unsigned short* Cbf = (unsigned short*)(ws + (1 << 20)); // B*288*2 ~ 75.5 MB

    zero_cnt_k<<<1, 32, 0, stream>>>(cnt);
    prep_w_k<<<(1024 * KP + 255) / 256, 256, 0, stream>>>(W_gates, Wbf);
    prep_combined_k<<<8192, 256, 0, stream>>>(input, h_cur, Cbf);
    switch_count_k<<<BROWS / 256, 256, 0, stream>>>(Cbf, W_switch, b_switch, cnt);

    const size_t lds_bytes = (size_t)(64 + 128) * KP * sizeof(unsigned short); // 110592
    lstm_wmma_k<<<BROWS / 64, 256, lds_bytes, stream>>>(Wbf, Cbf, b_gates, c_cur, cnt, out);
}
